// MultiHeadPerformerAttention_58351425683920
// MI455X (gfx1250) — matrix operand order repaired and checked by reading, not compiled, not run on hardware
//
#include <hip/hip_runtime.h>
#include <hip/hip_bf16.h>

typedef _Float16 F16;
typedef __attribute__((ext_vector_type(16))) _Float16 v16h;
typedef __attribute__((ext_vector_type(8)))  _Float16 h8;
typedef __attribute__((ext_vector_type(8)))  float    v8f;
typedef unsigned int u32;

#define Bz   4
#define Tz   4096
#define Dz   512
#define Hz   8
#define DHz  64
#define Mz   128
#define ROWS 16384      /* B*T   */
#define NR   131072     /* B*T*H */
#define CC   32         /* scan chunk */

#define DATA_NORM 0.35355339059327379f   /* 64^-0.25 */
#define RATIO     0.08838834764831845f   /* 1/sqrt(128) */
#define EPSF      1e-6f

#define WMMA(a, b, c) __builtin_amdgcn_wmma_f32_16x16x32_f16(false, (a), false, (b), (short)0, (c), false, false)

/* ---- CDNA5 async global->LDS (ASYNCcnt path, CDNA5 ISA) ---- */
__device__ __forceinline__ u32 lds_addr_of(const void* p) {
  /* generic LDS address keeps the LDS byte offset in addr[31:0] */
  return (u32)(unsigned long long)p;
}
__device__ __forceinline__ void async_ld_b128(u32 lds, u32 goff_bytes, const void* gbase) {
  asm volatile("global_load_async_to_lds_b128 %0, %1, %2"
               :: "v"(lds), "v"(goff_bytes), "s"(gbase) : "memory");
}
#define WAIT_ASYNC() asm volatile("s_wait_asynccnt 0x0" ::: "memory")

/* ---- fragment loaders per CDNA5 ISA 7.12.2 (wave32) ----
   A row-major and B column-major sources give each lane two contiguous
   16-byte runs -> ds_load_b128.                                          */
__device__ __forceinline__ v16h frag_a_row(const F16* s, int ld, int row0, int k0, int lane) {
  int m  = row0 + (lane & 15);
  int kb = k0 + ((lane & 16) ? 8 : 0);
  v16h a;
#pragma unroll
  for (int e = 0; e < 16; ++e) {
    int v = e >> 1;
    int k = kb + ((v & 4) ? 16 : 0) + 2 * (v & 3) + (e & 1);
    a[e] = s[m * ld + k];
  }
  return a;
}
/* A[m][k] where source is stored [k][m] (transposed read; scalar, used sparingly) */
__device__ __forceinline__ v16h frag_a_col(const F16* s, int ld, int row0, int k0, int lane) {
  int m  = row0 + (lane & 15);
  int kb = k0 + ((lane & 16) ? 8 : 0);
  v16h a;
#pragma unroll
  for (int e = 0; e < 16; ++e) {
    int v = e >> 1;
    int k = kb + ((v & 4) ? 16 : 0) + 2 * (v & 3) + (e & 1);
    a[e] = s[k * ld + m];
  }
  return a;
}
/* B[k][n] where source is stored column-major s[n*ld + k] (contiguous in k) */
__device__ __forceinline__ v16h frag_b_col(const F16* s, int ld, int k0, int col0, int lane) {
  int n  = col0 + (lane & 15);
  int kb = k0 + ((lane & 16) ? 8 : 0);
  v16h b;
#pragma unroll
  for (int e = 0; e < 16; ++e) b[e] = s[n * ld + kb + e + (e & 8)];
  return b;
}

__device__ __forceinline__ u32 f2ord(float f) {
  u32 u = __float_as_uint(f);
  return (u & 0x80000000u) ? ~u : (u | 0x80000000u);
}
__device__ __forceinline__ float ord2f(u32 u) {
  return __uint_as_float((u & 0x80000000u) ? (u & 0x7FFFFFFFu) : ~u);
}

/* ------------------- elementwise f32 -> f16 ------------------- */
__global__ void perf_cvt_f16(const float* __restrict__ in, F16* __restrict__ out, int n) {
  for (int i = blockIdx.x * blockDim.x + threadIdx.x; i < n; i += gridDim.x * blockDim.x)
    out[i] = (F16)in[i];
}
/* 512x512 f32 [k][n] -> f16 transposed [n][k] (for column-major B tiles) */
__global__ void perf_cvt_f16_t(const float* __restrict__ in, F16* __restrict__ out) {
  for (int i = blockIdx.x * blockDim.x + threadIdx.x; i < 512 * 512; i += gridDim.x * blockDim.x) {
    int k = i >> 9, n = i & 511;
    out[n * 512 + k] = (F16)in[i];
  }
}

/* ------------------- 512-K GEMM: [16384,512] x [512,512] + bias -------------------
   Wt pre-transposed [n][k]. Block tile 128x128, 8 waves, BK=64.
   Double-buffered async global->LDS staging (ASYNCcnt).                          */
__global__ __launch_bounds__(256) void perf_gemm512(const F16* __restrict__ A,
                                                    const F16* __restrict__ Wt,
                                                    const float* __restrict__ bias,
                                                    float* __restrict__ outF,
                                                    F16* __restrict__ outVT) {
  __shared__ F16 As[2][128][64];    /* [m][k] */
  __shared__ F16 Bs[2][128][64];    /* [n][k] column-major operand */
  const int tid = threadIdx.x, wave = tid >> 5, lane = tid & 31;
  const int br = blockIdx.x >> 2, bc = blockIdx.x & 3;
  const int row0 = br * 128, col0 = bc * 128;
  const int wr = wave & 3, wc = wave >> 2;
  const v8f vzero = {};
  v8f acc[2][4];
#pragma unroll
  for (int i = 0; i < 2; ++i)
#pragma unroll
    for (int j = 0; j < 4; ++j) acc[i][j] = vzero;

  /* async-stage one 128x64 A tile + 128x64 B tile: 4+4 b128 per thread */
  auto stage = [&](int k0, int buf) {
#pragma unroll
    for (int j = 0; j < 4; ++j) {
      int idx = tid + 256 * j;
      int r = idx >> 3, c = idx & 7;                 /* row, 8-half chunk */
      async_ld_b128(lds_addr_of(&As[buf][r][c * 8]),
                    (u32)(((row0 + r) * 512 + k0 + c * 8) * 2), A);
      async_ld_b128(lds_addr_of(&Bs[buf][r][c * 8]),
                    (u32)(((col0 + r) * 512 + k0 + c * 8) * 2), Wt);
    }
  };

  stage(0, 0);
  WAIT_ASYNC();
  __syncthreads();
  for (int it = 0; it < 8; ++it) {
    int buf = it & 1;
    if (it < 7) stage((it + 1) * 64, buf ^ 1);       /* overlap with compute */
#pragma unroll
    for (int ks = 0; ks < 64; ks += 32) {
      v16h bf[4];
#pragma unroll
      for (int j = 0; j < 4; ++j) bf[j] = frag_b_col(&Bs[buf][0][0], 64, ks, wc * 64 + j * 16, lane);
#pragma unroll
      for (int i = 0; i < 2; ++i) {
        v16h a = frag_a_row(&As[buf][0][0], 64, wr * 32 + i * 16, ks, lane);
#pragma unroll
        for (int j = 0; j < 4; ++j) acc[i][j] = WMMA(a, bf[j], acc[i][j]);
      }
    }
    if (it < 7) WAIT_ASYNC();
    __syncthreads();
  }
#pragma unroll
  for (int i = 0; i < 2; ++i)
#pragma unroll
    for (int j = 0; j < 4; ++j)
#pragma unroll
      for (int r = 0; r < 8; ++r) {
        int m = row0 + wr * 32 + i * 16 + (((lane & 16) ? 8 : 0) + r);
        int n = col0 + wc * 64 + j * 16 + (lane & 15);
        float v = acc[i][j][r] + bias[n];
        if (outF) outF[m * 512 + n] = v;
        if (outVT) {
          int hh = n >> 6, dd = n & 63, bb = m >> 12, tt = m & 4095;
          outVT[(((bb * 8 + hh) * 64 + dd) << 12) + tt] = (F16)v;
        }
      }
}

/* ------------------- FAVOR+ feature map -------------------
   mode 0: query (per-row stabilizer) / 1: key max pass / 2: key emit pass */
__global__ __launch_bounds__(256) void perf_features(const float* __restrict__ src,
                                                     const F16* __restrict__ projh,
                                                     F16* __restrict__ fout,
                                                     u32* __restrict__ stab,
                                                     int mode) {
  __shared__ F16 Ps[128][64];       /* proj [m][k] -> column-major B */
  __shared__ F16 Xs[8][16][64];
  __shared__ float Dg[8][16];
  const int tid = threadIdx.x, wave = tid >> 5, lane = tid & 31;
  const int rowbase = (blockIdx.x * 8 + wave) * 16;

  { /* async-stage proj: 4 b128 per thread */
#pragma unroll
    for (int j = 0; j < 4; ++j) {
      int idx = tid + 256 * j;          /* 1024 chunks of 8 halves */
      async_ld_b128(lds_addr_of(&Ps[0][0] + idx * 8), (u32)(idx * 16), projh);
    }
  }
  { /* stage xp + diag: lane -> row (lane&15), k-half */
    int rr = lane & 15, kh = (lane & 16) ? 32 : 0;
    const float4* s4 = (const float4*)(src + (rowbase + rr) * 64 + kh);
    float ss = 0.f;
#pragma unroll
    for (int j = 0; j < 4; ++j) {
      float4 x0 = s4[2 * j], x1 = s4[2 * j + 1];
      float xv[8] = {x0.x, x0.y, x0.z, x0.w, x1.x, x1.y, x1.z, x1.w};
      h8 p;
#pragma unroll
      for (int e = 0; e < 8; ++e) {
        float x = xv[e] * DATA_NORM;
        p[e] = (F16)x;
        ss += x * x;
      }
      *(h8*)&Xs[wave][rr][kh + 8 * j] = p;
    }
    ss += __shfl_xor(ss, 16, 32);
    if (lane < 16) Dg[wave][rr] = 0.5f * ss;
  }
  WAIT_ASYNC();
  __syncthreads();

  const v8f vzero = {};
  v8f acc[8];
#pragma unroll
  for (int t = 0; t < 8; ++t) acc[t] = vzero;
#pragma unroll
  for (int ks = 0; ks < 64; ks += 32) {
    v16h a = frag_a_row(&Xs[wave][0][0], 64, 0, ks, lane);
#pragma unroll
    for (int t = 0; t < 8; ++t) {
      v16h b = frag_b_col(&Ps[0][0], 64, ks, t * 16, lane);
      acc[t] = WMMA(a, b, acc[t]);
    }
  }

  float stabr[8];
#pragma unroll
  for (int r = 0; r < 8; ++r) {
    float mx = acc[0][r];
#pragma unroll
    for (int t = 1; t < 8; ++t) mx = fmaxf(mx, acc[t][r]);
    mx = fmaxf(mx, __shfl_xor(mx, 1, 32));
    mx = fmaxf(mx, __shfl_xor(mx, 2, 32));
    mx = fmaxf(mx, __shfl_xor(mx, 4, 32));
    mx = fmaxf(mx, __shfl_xor(mx, 8, 32));
    stabr[r] = mx;
  }

  if (mode == 0) {
#pragma unroll
    for (int t = 0; t < 8; ++t)
#pragma unroll
      for (int r = 0; r < 8; ++r) {
        int m = ((lane & 16) ? 8 : 0) + r;
        float v = RATIO * (expf(acc[t][r] - Dg[wave][m] - stabr[r]) + EPSF);
        fout[(rowbase + m) * 128 + t * 16 + (lane & 15)] = (F16)v;
      }
  } else if (mode == 1) {
    if ((lane & 15) == 0) {
      int mb = (lane & 16) ? 8 : 0;
#pragma unroll
      for (int r = 0; r < 8; ++r) {
        int row = rowbase + mb + r;
        atomicMax(stab + ((row >> 15) * 8 + (row & 7)), f2ord(stabr[r]));
      }
    }
  } else {
#pragma unroll
    for (int t = 0; t < 8; ++t)
#pragma unroll
      for (int r = 0; r < 8; ++r) {
        int m = ((lane & 16) ? 8 : 0) + r;
        int row = rowbase + m;
        float st = ord2f(stab[(row >> 15) * 8 + (row & 7)]);
        float v = RATIO * (expf(acc[t][r] - Dg[wave][m] - st) + EPSF);
        fout[row * 128 + t * 16 + (lane & 15)] = (F16)v;
      }
  }
}

/* ------------------- chunked causal linear-attention scan -------------------
   1 block per (b,h); KV state in WMMA accumulators. Next chunk's Q/K/V are
   async-prefetched into the alternate LDS buffer while this chunk computes.  */
__global__ __launch_bounds__(256) void perf_scan(const F16* __restrict__ qf,
                                                 const F16* __restrict__ kf,
                                                 const F16* __restrict__ vt,
                                                 F16* __restrict__ attn) {
  __shared__ F16 Qs[2][CC][128];  /* [c][m] */
  __shared__ F16 Ks[2][CC][128];  /* [c][m] */
  __shared__ F16 Vs[2][64][CC];   /* [d][c] column-major for B */
  __shared__ F16 KVh[64][128];    /* [d][m] column-major for B */
  __shared__ F16 Sm[CC][CC];      /* [c][c'] row-major A */
  __shared__ float ksum[128];
  __shared__ float den[CC];
  const int bh = blockIdx.x, b = bh >> 3, h = bh & 7;
  const int tid = threadIdx.x, wave = tid >> 5, lane = tid & 31;
  const v8f vzero = {};
  v8f kv[4];
#pragma unroll
  for (int d = 0; d < 4; ++d) kv[d] = vzero;
  if (tid < 128) ksum[tid] = 0.f;

  /* async-stage chunk t0 into buffer buf: 2+2+1 b128 per thread */
  auto stageQKV = [&](int t0, int buf) {
#pragma unroll
    for (int j = 0; j < 2; ++j) {
      int idx = tid + 256 * j;          /* 512 chunks: c = idx>>4, u = idx&15 */
      int c = idx >> 4, u = idx & 15;
      u32 goff = (u32)((((b * Tz + t0 + c) * 8 + h) * 128 + u * 8) * 2);
      async_ld_b128(lds_addr_of(&Qs[buf][c][u * 8]), goff, qf);
      async_ld_b128(lds_addr_of(&Ks[buf][c][u * 8]), goff, kf);
    }
    {
      int d = tid >> 2, j = tid & 3;    /* 256 chunks */
      async_ld_b128(lds_addr_of(&Vs[buf][d][j * 8]),
                    (u32)((((bh * 64 + d) << 12) + t0 + j * 8) * 2), vt);
    }
  };

  stageQKV(0, 0);
  WAIT_ASYNC();
  __syncthreads();

  for (int t0 = 0; t0 < Tz; t0 += CC) {
    const int buf = (t0 / CC) & 1;
    const int pf = (t0 + CC < Tz);
    if (pf) stageQKV(t0 + CC, buf ^ 1);            /* overlap with this chunk */

    /* stage KV state (pre-chunk) to LDS f16 [d][m]: 8 consecutive m per store */
#pragma unroll
    for (int dt = 0; dt < 4; ++dt) {
      h8 p;
#pragma unroll
      for (int r = 0; r < 8; ++r) p[r] = (F16)kv[dt][r];
      int d = dt * 16 + (lane & 15);
      int mb = wave * 16 + ((lane & 16) ? 8 : 0);
      *(h8*)&KVh[d][mb] = p;
    }
    __syncthreads();

    /* S = Qf x Kf^T, causal-masked (diag incl.) */
    if (wave < 4) {
      int ci = wave >> 1, cj = wave & 1;
      v8f s = vzero;
#pragma unroll
      for (int ks = 0; ks < 128; ks += 32) {
        v16h a = frag_a_row(&Qs[buf][0][0], 128, ci * 16, ks, lane);
        v16h bb = frag_b_col(&Ks[buf][0][0], 128, ks, cj * 16, lane);
        s = WMMA(a, bb, s);
      }
#pragma unroll
      for (int r = 0; r < 8; ++r) {
        int cm = ci * 16 + (((lane & 16) ? 8 : 0) + r);
        int cn = cj * 16 + (lane & 15);
        Sm[cm][cn] = (cn <= cm) ? (F16)s[r] : (F16)0.f;
      }
    }
    __syncthreads();

    /* den (uses pre-chunk ksum) */
    if (tid < CC) {
      float s1 = 0.f;
      for (int c2 = 0; c2 <= tid; ++c2) s1 += (float)Sm[tid][c2];
      for (int m = 0; m < 128; ++m) s1 += (float)Qs[buf][tid][m] * ksum[m];
      den[tid] = s1;
    }
    /* num = Smask x V + Qf x KV */
    v8f nm = vzero;
    {
      int ci = wave >> 2, dj = wave & 3;
      v16h a = frag_a_row(&Sm[0][0], CC, ci * 16, 0, lane);
      v16h bb = frag_b_col(&Vs[buf][0][0], CC, 0, dj * 16, lane);
      nm = WMMA(a, bb, nm);
#pragma unroll
      for (int ks = 0; ks < 128; ks += 32) {
        a = frag_a_row(&Qs[buf][0][0], 128, ci * 16, ks, lane);
        bb = frag_b_col(&KVh[0][0], 128, ks, dj * 16, lane);
        nm = WMMA(a, bb, nm);
      }
    }
    __syncthreads();

    { /* attn = num / den -> f16 (row-major [bt][h*64+d]) */
      int ci = wave >> 2, dj = wave & 3;
#pragma unroll
      for (int r = 0; r < 8; ++r) {
        int cm = ci * 16 + (((lane & 16) ? 8 : 0) + r);
        int n = dj * 16 + (lane & 15);
        int grow = (b * Tz + t0 + cm) * 8 + h;
        attn[grow * 64 + n] = (F16)(nm[r] / den[cm]);
      }
    }
    /* KV += Kf^T x V (wave w owns m-rows 16w..16w+15) */
    {
      v16h aT = frag_a_col(&Ks[buf][0][0], 128, wave * 16, 0, lane);
#pragma unroll
      for (int dt = 0; dt < 4; ++dt) {
        v16h bb = frag_b_col(&Vs[buf][0][0], CC, 0, dt * 16, lane);
        kv[dt] = WMMA(aT, bb, kv[dt]);
      }
    }
    /* ksum += colsum(Kf) */
    if (tid < 128) {
      float s = ksum[tid];
      for (int c = 0; c < CC; ++c) s += (float)Ks[buf][c][tid];
      ksum[tid] = s;
    }
    if (pf) WAIT_ASYNC();
    __syncthreads();
  }
}

extern "C" void kernel_launch(void* const* d_in, const int* in_sizes, int n_in,
                              void* d_out, int out_size, void* d_ws, size_t ws_size,
                              hipStream_t stream) {
  (void)in_sizes; (void)n_in; (void)out_size; (void)ws_size;
  const float* query = (const float*)d_in[0];
  const float* key_  = (const float*)d_in[1];
  const float* value = (const float*)d_in[2];
  const float* wq = (const float*)d_in[3];
  const float* bq = (const float*)d_in[4];
  const float* wk = (const float*)d_in[5];
  const float* bk = (const float*)d_in[6];
  const float* wv = (const float*)d_in[7];
  const float* bv = (const float*)d_in[8];
  const float* wo = (const float*)d_in[9];
  const float* bo = (const float*)d_in[10];
  const float* proj = (const float*)d_in[11];

  char* base = (char*)d_ws;
  size_t off = 0;
  auto take = [&](size_t bytes) { size_t r = off; off += (bytes + 255) & ~size_t(255); return r; };
  F16*   xh    = (F16*)(base + take((size_t)ROWS * 512 * 2));   /* reused as attn */
  F16*   wbuf  = (F16*)(base + take((size_t)512 * 512 * 2));    /* transposed weights */
  F16*   projh = (F16*)(base + take((size_t)Mz * DHz * 2));
  float* qproj = (float*)(base + take((size_t)ROWS * 512 * 4)); /* reused as kf */
  float* kproj = (float*)(base + take((size_t)ROWS * 512 * 4));
  F16*   vt    = (F16*)(base + take((size_t)ROWS * 512 * 2));   /* V transposed [b,h,d,t] */
  F16*   qf    = (F16*)(base + take((size_t)NR * Mz * 2));
  u32*   stab  = (u32*)(base + take(32 * sizeof(u32)));
  F16*   kf    = (F16*)qproj;
  F16*   attn  = xh;

  hipMemsetAsync(stab, 0, 32 * sizeof(u32), stream);
  perf_cvt_f16<<<32, 256, 0, stream>>>(proj, projh, Mz * DHz);

  /* Q projection */
  perf_cvt_f16<<<4096, 256, 0, stream>>>(query, xh, ROWS * Dz);
  perf_cvt_f16_t<<<256, 256, 0, stream>>>(wq, wbuf);
  perf_gemm512<<<512, 256, 0, stream>>>(xh, wbuf, bq, qproj, (F16*)nullptr);
  /* K projection */
  perf_cvt_f16<<<4096, 256, 0, stream>>>(key_, xh, ROWS * Dz);
  perf_cvt_f16_t<<<256, 256, 0, stream>>>(wk, wbuf);
  perf_gemm512<<<512, 256, 0, stream>>>(xh, wbuf, bk, kproj, (F16*)nullptr);
  /* V projection -> transposed f16 [b,h,d,t] */
  perf_cvt_f16<<<4096, 256, 0, stream>>>(value, xh, ROWS * Dz);
  perf_cvt_f16_t<<<256, 256, 0, stream>>>(wv, wbuf);
  perf_gemm512<<<512, 256, 0, stream>>>(xh, wbuf, bv, (float*)nullptr, vt);

  /* features */
  perf_features<<<1024, 256, 0, stream>>>(qproj, projh, qf, stab, 0);
  perf_features<<<1024, 256, 0, stream>>>(kproj, projh, (F16*)nullptr, stab, 1);
  perf_features<<<1024, 256, 0, stream>>>(kproj, projh, kf, stab, 2);

  /* chunked causal scan with async double-buffering */
  perf_scan<<<32, 256, 0, stream>>>(qf, kf, vt, attn);

  /* output projection */
  perf_cvt_f16_t<<<256, 256, 0, stream>>>(wo, wbuf);
  perf_gemm512<<<512, 256, 0, stream>>>(attn, wbuf, bo, (float*)d_out, (F16*)nullptr);
}
